// SSD_two_stage_end2end_44074954392074
// MI455X (gfx1250) — matrix operand order repaired and checked by reading, not compiled, not run on hardware
//
#include <hip/hip_runtime.h>
#include <hip/hip_bf16.h>

typedef __attribute__((ext_vector_type(16))) _Float16 v16h;
typedef __attribute__((ext_vector_type(8)))  _Float16 v8h;
typedef __attribute__((ext_vector_type(8)))  float    v8f;

// ---------------------------------------------------------------------------
// Workspace layout (bytes)
// ---------------------------------------------------------------------------
static constexpr size_t OFF_RPN   = 0;                                   // 200*10 f32
static constexpr size_t OFF_LP    = 8192;                                // 8*4 f32
static constexpr size_t OFF_VMASK = 8448;                                // 8 f32
static constexpr size_t OFF_POS   = 8704;                                // 8*13 f32
static constexpr size_t OFF_BH    = 9216;                                // 3*16 f32 fused head bias
static constexpr size_t OFF_CROPS = 16384;                               // 8*64*56*56 f16
static constexpr size_t OFF_W0    = OFF_CROPS + 8ull*64*56*56*2;         // 9*512*64 f16
static constexpr size_t OFF_WM    = OFF_W0 + 9ull*512*64*2;              // 5*9*512*512 f16
static constexpr size_t OFF_WH    = OFF_WM + 5ull*9*512*512*2;           // 3*9*16*512 f16
static constexpr size_t OFF_A     = OFF_WH + 3ull*9*16*512*2;            // 8*512*56*56 f16
static constexpr size_t OFF_B     = OFF_A + 8ull*512*56*56*2;            // 8*512*56*56 f16
static constexpr size_t OFF_H0    = OFF_B + 8ull*512*56*56*2;            // 8*16*3136 f32
static constexpr size_t OFF_H1    = OFF_H0 + 8ull*16*3136*4;             // 8*16*784  f32
static constexpr size_t OFF_H2    = OFF_H1 + 8ull*16*784*4;              // 8*16*196  f32

// ===========================================================================
// Stage 1: scores -> top200 -> decode -> NMS -> compact -> ROI select -> lp
// ===========================================================================
__global__ __launch_bounds__(256) void k_stage1(
    const float* __restrict__ loc, const float* __restrict__ conf,
    const float* __restrict__ has_lp, const float* __restrict__ size_lp,
    const float* __restrict__ offset, const float* __restrict__ priors,
    float* __restrict__ rpn, float* __restrict__ lp_out, float* __restrict__ vmask_out)
{
    const int tid = threadIdx.x;
    __shared__ float ssc[8732];
    __shared__ float stv[200];
    __shared__ int   sti[200];
    __shared__ float stb[200 * 4];
    __shared__ float ste[200 * 5];
    __shared__ int   skeep[200];
    __shared__ float srpn[2000];
    __shared__ float rv[256];
    __shared__ int   ri[256];

    for (int p = tid; p < 8732; p += 256) {
        float d = conf[2 * p + 1] - conf[2 * p + 0];
        float s = 1.f / (1.f + expf(-d));
        ssc[p] = (s > 0.01f) ? s : 0.f;
    }
    __syncthreads();

    // iterative top-200 (lowest-index tie break, consumed entries marked -1)
    for (int it = 0; it < 200; ++it) {
        float bv = -2.f; int bi = 0x7fffffff;
        for (int p = tid; p < 8732; p += 256) {
            float v = ssc[p];
            if (v > bv || (v == bv && p < bi)) { bv = v; bi = p; }
        }
        rv[tid] = bv; ri[tid] = bi; __syncthreads();
        for (int s = 128; s > 0; s >>= 1) {
            if (tid < s) {
                if (rv[tid + s] > rv[tid] ||
                    (rv[tid + s] == rv[tid] && ri[tid + s] < ri[tid])) {
                    rv[tid] = rv[tid + s]; ri[tid] = ri[tid + s];
                }
            }
            __syncthreads();
        }
        if (tid == 0) { stv[it] = rv[0]; sti[it] = ri[0]; ssc[ri[0]] = -1.f; }
        __syncthreads();
    }

    // decode + extras for selected
    if (tid < 200) {
        int idx = sti[tid];
        float v = stv[tid]; if (v < 0.f) v = 0.f; stv[tid] = v;
        float p0 = priors[4 * idx + 0], p1 = priors[4 * idx + 1];
        float p2 = priors[4 * idx + 2], p3 = priors[4 * idx + 3];
        float l0 = loc[4 * idx + 0], l1 = loc[4 * idx + 1];
        float l2 = loc[4 * idx + 2], l3 = loc[4 * idx + 3];
        float cx = p0 + l0 * 0.1f * p2, cy = p1 + l1 * 0.1f * p3;
        float w = p2 * expf(l2 * 0.2f), h = p3 * expf(l3 * 0.2f);
        stb[tid * 4 + 0] = cx - 0.5f * w; stb[tid * 4 + 1] = cy - 0.5f * h;
        stb[tid * 4 + 2] = cx + 0.5f * w; stb[tid * 4 + 3] = cy + 0.5f * h;
        float hl = 1.f / (1.f + expf(-has_lp[idx]));
        ste[tid * 5 + 0] = hl;
        ste[tid * 5 + 1] = size_lp[2 * idx + 0];
        ste[tid * 5 + 2] = size_lp[2 * idx + 1];
        ste[tid * 5 + 3] = offset[2 * idx + 0];
        ste[tid * 5 + 4] = offset[2 * idx + 1];
        skeep[tid] = (v > 0.f) ? 1 : 0;
    }
    __syncthreads();

    // greedy NMS
    for (int i = 0; i < 199; ++i) {
        int ki = skeep[i];
        if (ki && tid > i && tid < 200 && skeep[tid]) {
            float ax1 = stb[i * 4 + 0], ay1 = stb[i * 4 + 1];
            float ax2 = stb[i * 4 + 2], ay2 = stb[i * 4 + 3];
            float bx1 = stb[tid * 4 + 0], by1 = stb[tid * 4 + 1];
            float bx2 = stb[tid * 4 + 2], by2 = stb[tid * 4 + 3];
            float ix1 = fmaxf(ax1, bx1), iy1 = fmaxf(ay1, by1);
            float ix2 = fminf(ax2, bx2), iy2 = fminf(ay2, by2);
            float inter = fmaxf(ix2 - ix1, 0.f) * fmaxf(iy2 - iy1, 0.f);
            float a = (ax2 - ax1) * (ay2 - ay1);
            float b = (bx2 - bx1) * (by2 - by1);
            float iou = inter / (a + b - inter + 1e-9f);
            if (iou > 0.45f) skeep[tid] = 0;
        }
        __syncthreads();
    }

    for (int e = tid; e < 2000; e += 256) srpn[e] = 0.f;
    __syncthreads();

    if (tid == 0) {
        int cnt = 0;
        for (int i = 0; i < 200; ++i) {
            if (skeep[i]) {
                srpn[cnt * 10 + 0] = stv[i];
                for (int q = 0; q < 4; ++q) srpn[cnt * 10 + 1 + q] = stb[i * 4 + q];
                for (int q = 0; q < 5; ++q) srpn[cnt * 10 + 5 + q] = ste[i * 5 + q];
                ++cnt;
            }
        }
        // top-8 ROI: valid rows (sorted desc already) first, then lowest-index -1 keys
        int topi[8]; float topv[8]; int c8 = 0;
        for (int i = 0; i < 200 && c8 < 8; ++i) {
            float sc = srpn[i * 10 + 0];
            if (sc > 0.6f && srpn[i * 10 + 5] > 0.5f) { topi[c8] = i; topv[c8] = sc; ++c8; }
        }
        for (int i = 0; i < 200 && c8 < 8; ++i) {
            float sc = srpn[i * 10 + 0];
            if (!(sc > 0.6f && srpn[i * 10 + 5] > 0.5f)) { topi[c8] = i; topv[c8] = -1.f; ++c8; }
        }
        for (int s = 0; s < 8; ++s) {
            const float* m = &srpn[topi[s] * 10];
            float ccx = 0.5f * (m[1] + m[3]), ccy = 0.5f * (m[2] + m[4]);
            float lcx = ccx + m[8], lcy = ccy + m[9];
            float hx = m[6], hy = m[7];   // 0.5 * size_lp * EXPAND(2.0)
            float x1 = lcx - hx, y1 = lcy - hy, x2 = lcx + hx, y2 = lcy + hy;
            x1 = fminf(fmaxf(x1, 0.f), 1.f); y1 = fminf(fmaxf(y1, 0.f), 1.f);
            x2 = fminf(fmaxf(x2, 0.f), 1.f); y2 = fminf(fmaxf(y2, 0.f), 1.f);
            x1 = fmaxf(x1, m[1]); y1 = fmaxf(y1, m[2]);
            x2 = fmaxf(x2, m[1]); y2 = fmaxf(y2, m[2]);
            x1 = fminf(x1, m[3]); y1 = fminf(y1, m[4]);
            x2 = fminf(x2, m[3]); y2 = fminf(y2, m[4]);
            lp_out[s * 4 + 0] = x1; lp_out[s * 4 + 1] = y1;
            lp_out[s * 4 + 2] = x2; lp_out[s * 4 + 3] = y2;
            vmask_out[s] = (topv[s] > 0.f) ? 1.f : 0.f;
        }
    }
    __syncthreads();
    for (int e = tid; e < 2000; e += 256) rpn[e] = srpn[e];
}

// ===========================================================================
// Weight prep: f32 OIHW -> f16 [kpos][co][ci]
// ===========================================================================
__global__ void k_prep_w0(const float* __restrict__ src, _Float16* __restrict__ dst)
{
    int idx = blockIdx.x * 256 + threadIdx.x;
    if (idx >= 512 * 64 * 9) return;
    int kpos = idx % 9, ci = (idx / 9) % 64, co = idx / 576;
    dst[(size_t)kpos * 512 * 64 + co * 64 + ci] = (_Float16)src[idx];
}

__global__ void k_prep_wmain(const float* __restrict__ src, _Float16* __restrict__ dst)
{
    int idx = blockIdx.x * 256 + threadIdx.x;
    if (idx >= 5 * 512 * 512 * 9) return;
    int kpos = idx % 9, ci = (idx / 9) % 512, co = (idx / 4608) % 512, l = idx / 2359296;
    dst[(size_t)l * 2359296 + (size_t)kpos * 262144 + co * 512 + ci] = (_Float16)src[idx];
}

__global__ void k_prep_whead(const float* __restrict__ locw, const float* __restrict__ confw,
                             const float* __restrict__ fcw, _Float16* __restrict__ dst)
{
    int idx = blockIdx.x * 256 + threadIdx.x;
    if (idx >= 3 * 16 * 512 * 9) return;
    int ci = idx % 512, co = (idx / 512) % 16, kpos = (idx / 8192) % 9, s = idx / 73728;
    float v = 0.f;
    if (co < 4)       v = locw[((size_t)(s * 4 + co) * 512 + ci) * 9 + kpos];
    else if (co < 6)  v = confw[((size_t)(s * 2 + (co - 4)) * 512 + ci) * 9 + kpos];
    else if (co < 14) v = fcw[((size_t)(s * 8 + (co - 6)) * 512 + ci) * 9 + kpos];
    dst[(size_t)s * 73728 + (size_t)kpos * 8192 + co * 512 + ci] = (_Float16)v;
}

__global__ void k_prep_bh(const float* __restrict__ lb, const float* __restrict__ cb,
                          const float* __restrict__ fb, float* __restrict__ dst)
{
    int idx = threadIdx.x;
    if (idx >= 48) return;
    int s = idx / 16, co = idx % 16;
    float v = 0.f;
    if (co < 4)       v = lb[s * 4 + co];
    else if (co < 6)  v = cb[s * 2 + (co - 4)];
    else if (co < 14) v = fb[s * 8 + (co - 6)];
    dst[idx] = v;
}

// ===========================================================================
// Crop & resize (bilinear) -> f16 crops [8][64][56][56]
// ===========================================================================
__global__ void k_crop(const float* __restrict__ feat, const float* __restrict__ lp,
                       _Float16* __restrict__ crops)
{
    int o = blockIdx.x * 256 + threadIdx.x;
    if (o >= 8 * 64 * 56 * 56) return;
    int j = o % 56, i = (o / 56) % 56, c = (o / 3136) % 64, k = o / (3136 * 64);
    float bx1 = lp[k * 4 + 0], by1 = lp[k * 4 + 1];
    float bx2 = lp[k * 4 + 2], by2 = lp[k * 4 + 3];
    float ri = (float)i * (1.f / 55.f), rj = (float)j * (1.f / 55.f);
    float ys = (by1 + (by2 - by1) * ri) * 299.f;
    float xs = (bx1 + (bx2 - bx1) * rj) * 299.f;
    float yf = fminf(fmaxf(floorf(ys), 0.f), 299.f);
    float xf = fminf(fmaxf(floorf(xs), 0.f), 299.f);
    int y0 = (int)yf, x0 = (int)xf;
    int y1 = min(y0 + 1, 299), x1 = min(x0 + 1, 299);
    float wy = ys - yf, wx = xs - xf;
    const float* fc = feat + (size_t)c * 300 * 300;
    float f00 = fc[y0 * 300 + x0], f01 = fc[y0 * 300 + x1];
    float f10 = fc[y1 * 300 + x0], f11 = fc[y1 * 300 + x1];
    float top = f00 * (1.f - wx) + f01 * wx;
    float bot = f10 * (1.f - wx) + f11 * wx;
    crops[o] = (_Float16)(top * (1.f - wy) + bot * wy);
}

// ===========================================================================
// Implicit-GEMM 3x3 SAME conv via V_WMMA_F32_16X16X32_F16.
// Weights: f16 [kpos][COUT][CIN].  Acts: f16 [img][C][H][W].
// Block = 256 thr = 8 waves (WAVES_C x WAVES_P).
// Each wave owns an MTILES x NTILES grid of 16x16 output tiles
// (MTILES co-tiles x NTILES pixel rows): A/B fragments are reused across
// the tile grid and the MTILES*NTILES accumulation chains are independent,
// letting the scheduler overlap WMMAs instead of serializing on one C.
// ===========================================================================
template<int CIN, int COUT, int HW, int WAVES_C, int WAVES_P, int MTILES, int NTILES,
         bool F16RELU>
__global__ __launch_bounds__(256) void k_conv3(
    const _Float16* __restrict__ in, const _Float16* __restrict__ wgt,
    const float* __restrict__ bias, void* __restrict__ outv)
{
    constexpr int TILE_Y = WAVES_P * NTILES;
    constexpr int HALO_R = TILE_Y + 2;
    constexpr int LDSN = HALO_R * 18 * 32;
    __shared__ __align__(64) _Float16 lds[LDSN];

    const int tid = threadIdx.x;
    const int lane = tid & 31;
    const int wv = tid >> 5;
    const int wc = wv % WAVES_C;
    const int wp = wv / WAVES_C;
    const int n16 = lane & 15;
    const int hi = lane >> 4;            // 0 or 1

    const int x0 = blockIdx.x * 16;
    const int y0 = blockIdx.y * TILE_Y;
    constexpr int CO_PER_BLOCK = 16 * WAVES_C * MTILES;
    constexpr int CG = COUT / CO_PER_BLOCK;
    const int img = blockIdx.z / CG;
    const int cg  = blockIdx.z % CG;
    const int co_base = cg * CO_PER_BLOCK + wc * (16 * MTILES);
    const int yb = y0 + wp * NTILES;     // first row owned by this wave
    const size_t in_img = (size_t)img * CIN * HW * HW;

    v8f acc[MTILES][NTILES];
#pragma unroll
    for (int t = 0; t < MTILES; ++t)
#pragma unroll
        for (int n = 0; n < NTILES; ++n) acc[t][n] = (v8f){};

    for (int ci0 = 0; ci0 < CIN; ci0 += 32) {
        // Stage input halo tile to LDS as [pos][ci] (pos = lr*18+lc)
        for (int e = tid; e < LDSN; e += 256) {
            int ci = e & 31;
            int pos = e >> 5;
            int lc = pos % 18, lr = pos / 18;
            int gx = x0 - 1 + lc, gy = y0 - 1 + lr;
            _Float16 v = (_Float16)0.f;
            if ((unsigned)gx < (unsigned)HW && (unsigned)gy < (unsigned)HW)
                v = in[in_img + (size_t)(ci0 + ci) * HW * HW + gy * HW + gx];
            lds[pos * 32 + ci] = v;
        }
        __syncthreads();

#pragma unroll
        for (int kpos = 0; kpos < 9; ++kpos) {
            const int ky = kpos / 3, kx = kpos % 3;

            // A fragments: lanes 0-15 -> K 0..15; lanes 16-31 -> K 16..31 (contiguous ci)
            v16h a[MTILES];
#pragma unroll
            for (int t = 0; t < MTILES; ++t) {
                const _Float16* wb = wgt
                    + ((size_t)kpos * COUT + (co_base + t * 16 + n16)) * CIN
                    + ci0 + hi * 16;
                __builtin_prefetch(wb + (size_t)COUT * CIN, 0, 1);   // next kpos block
                v8h a_lo = *(const v8h*)(wb);
                v8h a_hi = *(const v8h*)(wb + 8);
#pragma unroll
                for (int q = 0; q < 8; ++q) { a[t][q] = a_lo[q]; a[t][q + 8] = a_hi[q]; }
            }

            // B fragments: lanes 0-15 -> K 0..15, lanes 16-31 -> K 16..31 (contiguous ci)
            v16h b[NTILES];
#pragma unroll
            for (int n = 0; n < NTILES; ++n) {
                const int pos = (wp * NTILES + n + ky) * 18 + (n16 + kx);
                b[n] = *(const v16h*)(&lds[pos * 32 + hi * 16]);
            }

#pragma unroll
            for (int t = 0; t < MTILES; ++t)
#pragma unroll
                for (int n = 0; n < NTILES; ++n)
                    acc[t][n] = __builtin_amdgcn_wmma_f32_16x16x32_f16(
                        false, a[t], false, b[n], (short)0, acc[t][n], false, false);
        }
        __syncthreads();
    }

    const int x = x0 + n16;
#pragma unroll
    for (int t = 0; t < MTILES; ++t) {
#pragma unroll
        for (int n = 0; n < NTILES; ++n) {
            const int y = yb + n;
            if (x < HW && y < HW) {
#pragma unroll
                for (int r = 0; r < 8; ++r) {
                    int m = r + hi * 8;      // C/D layout: lanes>=16 hold M=8..15
                    int co = co_base + t * 16 + m;
                    float v = acc[t][n][r] + bias[co];
                    size_t oidx = (size_t)img * COUT * HW * HW + (size_t)co * HW * HW
                                + (size_t)y * HW + x;
                    if (F16RELU) ((_Float16*)outv)[oidx] = (_Float16)fmaxf(v, 0.f);
                    else         ((float*)outv)[oidx]   = v;
                }
            }
        }
    }
}

// ===========================================================================
// 2x2 max pool, f16
// ===========================================================================
__global__ void k_pool(const _Float16* __restrict__ in, _Float16* __restrict__ out,
                       int H, int total)
{
    int o = blockIdx.x * 256 + threadIdx.x;
    if (o >= total) return;
    int W2 = H / 2;
    int j = o % W2, i = (o / W2) % W2, c = o / (W2 * W2);   // c = img*512+ch
    const _Float16* p = in + ((size_t)c * H + 2 * i) * H + 2 * j;
    float a = (float)p[0], b = (float)p[1], d = (float)p[H], e = (float)p[H + 1];
    out[o] = (_Float16)fmaxf(fmaxf(a, b), fmaxf(d, e));
}

// ===========================================================================
// Stage 2: per-ROI argmax (== row 0 of nms_rows) + decode + coord transform
// ===========================================================================
__global__ __launch_bounds__(256) void k_stage2(
    const float* __restrict__ H0, const float* __restrict__ H1, const float* __restrict__ H2,
    const float* __restrict__ priors2, const float* __restrict__ lp, float* __restrict__ pos)
{
    const int k = blockIdx.x;
    const int tid = threadIdx.x;
    __shared__ float sc[4116];
    __shared__ float rv[256];
    __shared__ int   ri[256];

    auto hp = [&](int ch, int p) -> float {
        if (p < 3136) return H0[((size_t)k * 16 + ch) * 3136 + p];
        if (p < 3920) return H1[((size_t)k * 16 + ch) * 784 + (p - 3136)];
        return H2[((size_t)k * 16 + ch) * 196 + (p - 3920)];
    };

    for (int p = tid; p < 4116; p += 256) {
        float d = hp(5, p) - hp(4, p);
        float s = 1.f / (1.f + expf(-d));
        sc[p] = (s > 0.01f) ? s : 0.f;
    }
    __syncthreads();

    float bv = -2.f; int bi = 0x7fffffff;
    for (int p = tid; p < 4116; p += 256) {
        float v = sc[p];
        if (v > bv || (v == bv && p < bi)) { bv = v; bi = p; }
    }
    rv[tid] = bv; ri[tid] = bi; __syncthreads();
    for (int s = 128; s > 0; s >>= 1) {
        if (tid < s) {
            if (rv[tid + s] > rv[tid] ||
                (rv[tid + s] == rv[tid] && ri[tid + s] < ri[tid])) {
                rv[tid] = rv[tid + s]; ri[tid] = ri[tid + s];
            }
        }
        __syncthreads();
    }

    if (tid == 0) {
        float val = rv[0]; int idx = ri[0];
        float row[13];
        for (int c = 0; c < 13; ++c) row[c] = 0.f;
        if (val > 0.f) {
            float p0 = priors2[4 * idx + 0], p1 = priors2[4 * idx + 1];
            float p2 = priors2[4 * idx + 2], p3 = priors2[4 * idx + 3];
            float l0 = hp(0, idx), l1 = hp(1, idx), l2 = hp(2, idx), l3 = hp(3, idx);
            float cx = p0 + l0 * 0.1f * p2, cy = p1 + l1 * 0.1f * p3;
            float w = p2 * expf(l2 * 0.2f), h = p3 * expf(l3 * 0.2f);
            row[0] = val;
            row[1] = cx - 0.5f * w; row[2] = cy - 0.5f * h;
            row[3] = cx + 0.5f * w; row[4] = cy + 0.5f * h;
            for (int j = 0; j < 8; ++j) {
                float base = (j & 1) ? p1 : p0;
                float scl  = (j & 1) ? p3 : p2;
                row[5 + j] = base + hp(6 + j, idx) * 0.1f * scl;
            }
        }
        // coords = row[1:13] * tile(size,6) + tile(tl,6)  (applied even to zero rows)
        float tx = lp[k * 4 + 0], ty = lp[k * 4 + 1];
        float sx = lp[k * 4 + 2] - tx, sy = lp[k * 4 + 3] - ty;
        pos[k * 13 + 0] = row[0];
        for (int j = 0; j < 12; ++j) {
            float mlt = (j & 1) ? sy : sx;
            float add = (j & 1) ? ty : tx;
            pos[k * 13 + 1 + j] = row[1 + j] * mlt + add;
        }
    }
}

// ===========================================================================
// Final assembly: out (1,3,200,13) f32
// ===========================================================================
__global__ void k_assemble(const float* __restrict__ rpn, const float* __restrict__ lp,
                           const float* __restrict__ vmask, const float* __restrict__ pos,
                           float* __restrict__ out)
{
    int o = blockIdx.x * 256 + threadIdx.x;
    if (o >= 3 * 200 * 13) return;
    int c = o % 13, r = (o / 13) % 200, pl = o / (13 * 200);
    float v = 0.f;
    if (pl == 1) {
        if (c < 5) v = rpn[r * 10 + c];
        else if (r < 8) {
            if (c < 9)       v = lp[r * 4 + (c - 5)] * vmask[r];
            else if (c == 9) v = vmask[r];
        }
    } else if (pl == 2) {
        if (r < 8) v = pos[r * 13 + c] * vmask[r];
    }
    out[o] = v;
}

// ===========================================================================
// Launch
// ===========================================================================
extern "C" void kernel_launch(void* const* d_in, const int* in_sizes, int n_in,
                              void* d_out, int out_size, void* d_ws, size_t ws_size,
                              hipStream_t stream) {
    const float* loc      = (const float*)d_in[0];
    const float* conf     = (const float*)d_in[1];
    const float* has_lp   = (const float*)d_in[2];
    const float* size_lp  = (const float*)d_in[3];
    const float* offset   = (const float*)d_in[4];
    const float* feat     = (const float*)d_in[5];
    const float* priors   = (const float*)d_in[6];
    const float* priors2  = (const float*)d_in[7];
    const float* vgg2_w0  = (const float*)d_in[8];
    const float* vgg2_w   = (const float*)d_in[9];
    const float* vgg2_b   = (const float*)d_in[10];
    const float* loc2_w   = (const float*)d_in[11];
    const float* loc2_b   = (const float*)d_in[12];
    const float* conf2_w  = (const float*)d_in[13];
    const float* conf2_b  = (const float*)d_in[14];
    const float* fc2_w    = (const float*)d_in[15];
    const float* fc2_b    = (const float*)d_in[16];
    float* out = (float*)d_out;

    char* ws = (char*)d_ws;
    float*    rpn   = (float*)(ws + OFF_RPN);
    float*    lp    = (float*)(ws + OFF_LP);
    float*    vmask = (float*)(ws + OFF_VMASK);
    float*    pos   = (float*)(ws + OFF_POS);
    float*    bh    = (float*)(ws + OFF_BH);
    _Float16* crops = (_Float16*)(ws + OFF_CROPS);
    _Float16* W0    = (_Float16*)(ws + OFF_W0);
    _Float16* WM    = (_Float16*)(ws + OFF_WM);
    _Float16* WH    = (_Float16*)(ws + OFF_WH);
    _Float16* actA  = (_Float16*)(ws + OFF_A);
    _Float16* actB  = (_Float16*)(ws + OFF_B);
    float*    H0    = (float*)(ws + OFF_H0);
    float*    H1    = (float*)(ws + OFF_H1);
    float*    H2    = (float*)(ws + OFF_H2);

    // Stage 1 (also produces lp boxes for crops)
    k_stage1<<<1, 256, 0, stream>>>(loc, conf, has_lp, size_lp, offset, priors,
                                    rpn, lp, vmask);

    // Weight relayout f32 -> f16 [kpos][co][ci]
    k_prep_w0<<<(512 * 64 * 9 + 255) / 256, 256, 0, stream>>>(vgg2_w0, W0);
    k_prep_wmain<<<(5 * 512 * 512 * 9 + 255) / 256, 256, 0, stream>>>(vgg2_w, WM);
    k_prep_whead<<<(3 * 16 * 512 * 9 + 255) / 256, 256, 0, stream>>>(loc2_w, conf2_w, fc2_w, WH);
    k_prep_bh<<<1, 64, 0, stream>>>(loc2_b, conf2_b, fc2_b, bh);

    // Crops (f16)
    k_crop<<<(8 * 64 * 56 * 56 + 255) / 256, 256, 0, stream>>>(feat, lp, crops);

    // VGG-2 stack: main convs use 2x2 register-blocked WMMA tiles
    // (block = 128 co x 4 rows x 16 px), heads use a single 16-ch tile.
    k_conv3<64, 512, 56, 4, 2, 2, 2, true><<<dim3(4, 14, 32), 256, 0, stream>>>(crops, W0, vgg2_b + 0, actA);
    k_conv3<512, 512, 56, 4, 2, 2, 2, true><<<dim3(4, 14, 32), 256, 0, stream>>>(actA, WM + 0ull * 2359296, vgg2_b + 512, actB);
    k_conv3<512, 16, 56, 1, 8, 1, 1, false><<<dim3(4, 7, 8), 256, 0, stream>>>(actB, WH + 0ull * 73728, bh + 0, H0);
    k_pool<<<(8 * 512 * 28 * 28 + 255) / 256, 256, 0, stream>>>(actB, actA, 56, 8 * 512 * 28 * 28);

    k_conv3<512, 512, 28, 4, 2, 2, 2, true><<<dim3(2, 7, 32), 256, 0, stream>>>(actA, WM + 1ull * 2359296, vgg2_b + 1024, actB);
    k_conv3<512, 512, 28, 4, 2, 2, 2, true><<<dim3(2, 7, 32), 256, 0, stream>>>(actB, WM + 2ull * 2359296, vgg2_b + 1536, actA);
    k_conv3<512, 16, 28, 1, 8, 1, 1, false><<<dim3(2, 4, 8), 256, 0, stream>>>(actA, WH + 1ull * 73728, bh + 16, H1);
    k_pool<<<(8 * 512 * 14 * 14 + 255) / 256, 256, 0, stream>>>(actA, actB, 28, 8 * 512 * 14 * 14);

    k_conv3<512, 512, 14, 4, 2, 2, 2, true><<<dim3(1, 4, 32), 256, 0, stream>>>(actB, WM + 3ull * 2359296, vgg2_b + 2048, actA);
    k_conv3<512, 512, 14, 4, 2, 2, 2, true><<<dim3(1, 4, 32), 256, 0, stream>>>(actA, WM + 4ull * 2359296, vgg2_b + 2560, actB);
    k_conv3<512, 16, 14, 1, 8, 1, 1, false><<<dim3(1, 2, 8), 256, 0, stream>>>(actB, WH + 2ull * 73728, bh + 32, H2);

    // Stage 2 + final assembly
    k_stage2<<<8, 256, 0, stream>>>(H0, H1, H2, priors2, lp, pos);
    k_assemble<<<(3 * 200 * 13 + 255) / 256, 256, 0, stream>>>(rpn, lp, vmask, pos, out);

    (void)in_sizes; (void)n_in; (void)out_size; (void)ws_size;
}
